// LocalAttention_65180423684704
// MI455X (gfx1250) — matrix operand order repaired and checked by reading, not compiled, not run on hardware
//
#include <hip/hip_runtime.h>
#include <math.h>

#define B_  8
#define S_  512
#define D_  768
#define H_  12
#define DK_ 64
#define NEGV 1.0e9f

typedef __attribute__((ext_vector_type(16))) _Float16 v16h;
typedef __attribute__((ext_vector_type(8)))  _Float16 h8;
typedef __attribute__((ext_vector_type(8)))  float    v8f;
typedef __attribute__((ext_vector_type(4)))  unsigned int v4u;
typedef __attribute__((ext_vector_type(8)))  int      v8i;
typedef __attribute__((ext_vector_type(4)))  int      v4i;

static __device__ inline v16h mk16(h8 lo, h8 hi) {
  v16h r;
#pragma unroll
  for (int i = 0; i < 8; ++i) { r[i] = lo[i]; r[8 + i] = hi[i]; }
  return r;
}

static __device__ inline float wred_max(float v) {
#pragma unroll
  for (int m = 16; m > 0; m >>= 1) v = fmaxf(v, __shfl_xor(v, m, 32));
  return v;
}
static __device__ inline float wred_sum(float v) {
#pragma unroll
  for (int m = 16; m > 0; m >>= 1) v += __shfl_xor(v, m, 32);
  return v;
}
static __device__ inline float wincl_scan(float v) {
  const int lane = threadIdx.x & 31;
#pragma unroll
  for (int o = 1; o < 32; o <<= 1) {
    float n = __shfl_up(v, o, 32);
    if (lane >= o) v += n;
  }
  return v;
}

// ---------------------------------------------------------------- packing ---
__global__ void cvt_f32_to_f16(const float* __restrict__ src,
                               _Float16* __restrict__ dst, int n) {
  int i = blockIdx.x * blockDim.x + threadIdx.x;
  if (i < n) dst[i] = (_Float16)src[i];
}

// Wt[n*D + k] = (f16) W[k*D + n]
__global__ void transpose_w_f16(const float* __restrict__ W,
                                _Float16* __restrict__ Wt) {
  int idx = blockIdx.x * blockDim.x + threadIdx.x;
  if (idx < D_ * D_) {
    int n = idx / D_, k = idx % D_;
    Wt[idx] = (_Float16)W[(size_t)k * D_ + n];
  }
}

// ------------------------------------------------------------- projections --
// grid (M/16, N/64, 6), block 128 (4 waves, each owns a 16-wide N sub-tile).
// The 16x768 A-tile of X is shared by all 4 waves -> staged once in LDS via
// the Tensor Data Mover (tensor_load_to_lds, TENSORcnt).
__global__ __launch_bounds__(128)
void proj_gemm(const _Float16* __restrict__ Xq, const _Float16* __restrict__ Xk,
               const _Float16* __restrict__ Wt6,
               const float* __restrict__ b0, const float* __restrict__ b1,
               const float* __restrict__ b2, const float* __restrict__ b3,
               const float* __restrict__ b4, const float* __restrict__ b5,
               _Float16* __restrict__ P6, float scale) {
  __shared__ _Float16 sX[16 * D_];   // 24 KB staged A-tile

  const int which = blockIdx.z;
  const _Float16* X  = (which & 1) ? Xk : Xq;
  const _Float16* Wt = Wt6 + (size_t)which * D_ * D_;
  const float* bias;
  switch (which) {
    case 0: bias = b0; break; case 1: bias = b1; break;
    case 2: bias = b2; break; case 3: bias = b3; break;
    case 4: bias = b4; break; default: bias = b5; break;
  }
  const float s = (which & 1) ? 1.0f : scale;
  _Float16* out = P6 + (size_t)which * (B_ * S_) * D_;

  const int lane = threadIdx.x & 31;
  const int wave = threadIdx.x >> 5;
  const int half = lane >> 4;
  const int l16  = lane & 15;
  const int m0   = blockIdx.x * 16;             // M tile base
  const int n0   = blockIdx.y * 64 + wave * 16; // this wave's N base

  // --- TDM: DMA the 16x768 f16 A-tile into LDS (one descriptor, wave 0).
  if (wave == 0) {
    const unsigned long long ga =
        (unsigned long long)(uintptr_t)(X + (size_t)m0 * D_);
    // D# group0: count=1, lds_addr, global_addr[56:0], type=2 (bits 127:126)
    v4u g0 = { 1u,
               (unsigned int)(uintptr_t)(&sX[0]),
               (unsigned int)ga,
               (unsigned int)((ga >> 32) & 0x01FFFFFFu) | 0x80000000u };
    // D# group1: data_size=2B; tensor_dim0=768, tensor_dim1=4096;
    //            tile_dim0=768, tile_dim1=16; tensor_dim0_stride=768
    v8i g1 = { 0x00010000,              // wg_mask=0 | data_size=1 (2B)
               0x03000000,              // tensor_dim0[15:0]=768 << 16
               0x10000000,              // tensor_dim0[31:16]=0 | dim1[15:0]=4096<<16
               0x03000000,              // dim1[31:16]=0 | tile_dim0=768 << 16
               16,                      // tile_dim1=16 | tile_dim2=0
               768,                     // tensor_dim0_stride[31:0]
               0, 0 };
    v4i zz = { 0, 0, 0, 0 };
    v8i z8 = { 0, 0, 0, 0, 0, 0, 0, 0 };
    __builtin_amdgcn_tensor_load_to_lds(g0, g1, zz, zz, z8, 0);
    __builtin_amdgcn_s_wait_tensorcnt(0);
  }
  __syncthreads();

  const _Float16* Xrow = &sX[0] + (size_t)l16 * D_;    // A: row m = lane%16
  const _Float16* Wrow = Wt + (size_t)(n0 + l16) * D_; // B: col n = lane%16

  v8f acc = {};
#pragma unroll 4
  for (int k0 = 0; k0 < D_; k0 += 32) {
    // A-layout (16x32 f16): lane(m,half): K chunks [half*8..+7], [16+half*8..+7]
    v16h a = mk16(*(const h8*)(Xrow + k0 + half * 8),
                  *(const h8*)(Xrow + k0 + 16 + half * 8));
    // B-layout (32x16 f16): lane(n,half): K chunks [half*8..+7], [16+half*8..+7]
    v16h b = mk16(*(const h8*)(Wrow + k0 + half * 8),
                  *(const h8*)(Wrow + k0 + 16 + half * 8));
    acc = __builtin_amdgcn_wmma_f32_16x16x32_f16(false, a, false, b,
                                                 (short)0, acc, false, false);
  }

  const int ncol = n0 + l16;
  const float bv = bias[ncol];
  const int hh = ncol / DK_, dd = ncol % DK_;
#pragma unroll
  for (int r = 0; r < 8; ++r) {
    const int mrow = m0 + r + 8 * half;   // C/D layout: VGPR r, half selects M+8
    const int bb = mrow >> 9, ss = mrow & 511;
    out[(((size_t)bb * H_ + hh) * S_ + ss) * DK_ + dd] =
        (_Float16)((acc[r] + bv) * s);
  }
}

// ---------------------------------------------------------------- attention --
__global__ void zero_loss(float* __restrict__ out) {
  out[(size_t)B_ * H_ * S_ * S_] = 0.0f;
}

// grid = B*H*(S/16), block 128 (4 waves). Dynamic LDS: 3*16*512 f32 + 4.
__global__ __launch_bounds__(128)
void attn_fused(const _Float16* __restrict__ P6, const float* __restrict__ mask,
                const float* __restrict__ span, float* __restrict__ out) {
  extern __shared__ float smem[];
  float* scL = smem;                  // [16][512]
  float* scR = smem + 16 * 512;
  float* scM = smem + 2 * 16 * 512;
  float* lbuf = smem + 3 * 16 * 512;  // [4]

  const int gid = blockIdx.x;
  const int qt = gid & 31;            // q tile (S/16 = 32)
  const int bh = gid >> 5;
  const int b  = bh / H_;
  const int h  = bh % H_;
  const size_t PBS = (size_t)(B_ * S_) * D_;

  const int lane = threadIdx.x & 31;
  const int wave = threadIdx.x >> 5;
  const int half = lane >> 4;
  const int l16  = lane & 15;

  // Prefetch this block's span rows while WMMA phase runs (global_prefetch_b8).
#pragma unroll
  for (int rr = 0; rr < 4; ++rr) {
    const int qg = qt * 16 + wave * 4 + rr;
    __builtin_prefetch(
        span + (((size_t)h * B_ + b) * S_ + qg) * S_ + lane * 16, 0, 1);
  }

  // --- Phase A: q-tile A operands (branches: 0=ql, 2=qr, 4=q), kept in regs.
  v16h qa0[3], qa1[3];
#pragma unroll
  for (int i = 0; i < 3; ++i) {
    const _Float16* qr =
        P6 + (size_t)(2 * i) * PBS + ((size_t)bh * S_ + qt * 16 + l16) * DK_;
    qa0[i] = mk16(*(const h8*)(qr + half * 8), *(const h8*)(qr + 16 + half * 8));
    qa1[i] = mk16(*(const h8*)(qr + 32 + half * 8), *(const h8*)(qr + 48 + half * 8));
  }

  // --- Phase B: scores via WMMA; each wave sweeps k-tiles kt = wave, wave+4,...
  for (int kt = wave; kt < 32; kt += 4) {
#pragma unroll
    for (int i = 0; i < 3; ++i) {
      const _Float16* kr =
          P6 + (size_t)(2 * i + 1) * PBS + ((size_t)bh * S_ + kt * 16 + l16) * DK_;
      v16h kb0 = mk16(*(const h8*)(kr + half * 8), *(const h8*)(kr + 16 + half * 8));
      v16h kb1 = mk16(*(const h8*)(kr + 32 + half * 8), *(const h8*)(kr + 48 + half * 8));
      v8f acc = {};
      acc = __builtin_amdgcn_wmma_f32_16x16x32_f16(false, qa0[i], false, kb0,
                                                   (short)0, acc, false, false);
      acc = __builtin_amdgcn_wmma_f32_16x16x32_f16(false, qa1[i], false, kb1,
                                                   (short)0, acc, false, false);
      float* dst = (i == 0) ? scL : (i == 1) ? scR : scM;
#pragma unroll
      for (int r = 0; r < 8; ++r)
        dst[(r + 8 * half) * 512 + kt * 16 + l16] = acc[r];
    }
  }
  __syncthreads();

  // --- Phase C: row-wise softmax / cumsum / masked softmax / BCE.
  float lacc = 0.0f;
#pragma unroll 1
  for (int rr = 0; rr < 4; ++rr) {
    const int qrow = wave * 4 + rr;
    const int qg = qt * 16 + qrow;
    const int c0 = lane * 16;
    float tl[16], tr[16], tm[16];
#pragma unroll
    for (int j = 0; j < 16; ++j) {
      tl[j] = scL[qrow * 512 + c0 + j];
      tr[j] = scR[qrow * 512 + c0 + j];
      tm[j] = scM[qrow * 512 + c0 + j];
    }

    // left branch: softmax(score - (k<=q ? 0 : 1e9)) then inclusive cumsum
    float mx = -3.0e38f;
#pragma unroll
    for (int j = 0; j < 16; ++j) {
      tl[j] -= ((c0 + j) <= qg) ? 0.0f : NEGV;
      mx = fmaxf(mx, tl[j]);
    }
    mx = wred_max(mx);
    float sum = 0.0f;
#pragma unroll
    for (int j = 0; j < 16; ++j) { tl[j] = __expf(tl[j] - mx); sum += tl[j]; }
    sum = wred_sum(sum);
    float inv = 1.0f / sum;
    float pref = 0.0f;
#pragma unroll
    for (int j = 0; j < 16; ++j) { pref += tl[j] * inv; tl[j] = pref; } // chunk incl
    {
      float tot = pref;
      float scan = wincl_scan(tot);
      float excl = scan - tot;
#pragma unroll
      for (int j = 0; j < 16; ++j) tl[j] += excl;   // sml = inclusive cumsum
    }

    // right branch: softmax(score - (k>=q ? 0 : 1e9)) then reverse cumsum
    mx = -3.0e38f;
#pragma unroll
    for (int j = 0; j < 16; ++j) {
      tr[j] -= ((c0 + j) >= qg) ? 0.0f : NEGV;
      mx = fmaxf(mx, tr[j]);
    }
    mx = wred_max(mx);
    sum = 0.0f;
#pragma unroll
    for (int j = 0; j < 16; ++j) { tr[j] = __expf(tr[j] - mx); sum += tr[j]; }
    sum = wred_sum(sum);
    inv = 1.0f / sum;
    pref = 0.0f;
#pragma unroll
    for (int j = 0; j < 16; ++j) {      // tr[j] -> chunk-exclusive prefix of theta
      float th = tr[j] * inv;
      tr[j] = pref;
      pref += th;
    }
    {
      float tot = pref;
      float scan = wincl_scan(tot);
      float excl = scan - tot;
      float total = __shfl(scan, 31, 32);
#pragma unroll
      for (int j = 0; j < 16; ++j) tr[j] = total - excl - tr[j]; // smr
    }

    // main branch: softmax((score - rm_q - rm_k) * soft_Mask)
    float mq = mask[(size_t)b * S_ + qg];
    mq = (mq == -10000.0f) ? NEGV : mq;
    mx = -3.0e38f;
#pragma unroll
    for (int j = 0; j < 16; ++j) {
      float mk = mask[(size_t)b * S_ + c0 + j];
      mk = (mk == -10000.0f) ? NEGV : mk;
      tm[j] = (tm[j] - mq - mk) * (tl[j] * tr[j]);
      mx = fmaxf(mx, tm[j]);
    }
    mx = wred_max(mx);
    sum = 0.0f;
#pragma unroll
    for (int j = 0; j < 16; ++j) { tm[j] = __expf(tm[j] - mx); sum += tm[j]; }
    sum = wred_sum(sum);
    inv = 1.0f / sum;

    float* orow = out + ((size_t)bh * S_ + qg) * S_ + c0;
    const float* srow = span + (((size_t)h * B_ + b) * S_ + qg) * S_ + c0;
#pragma unroll
    for (int j = 0; j < 16; ++j) {
      float p = tm[j] * inv;
      orow[j] = p;
      // BCE vs sigmoid(p):  log1p(exp(-p)) + (1-span)*p   (exact rewrite)
      float sp = srow[j];
      lacc += log1pf(__expf(-p)) + (1.0f - sp) * p;
    }
  }

  // block loss reduction -> atomic add of scaled partial
  lacc = wred_sum(lacc);
  if (lane == 0) lbuf[wave] = lacc;
  __syncthreads();
  if (threadIdx.x == 0) {
    float tot = lbuf[0] + lbuf[1] + lbuf[2] + lbuf[3];
    atomicAdd(out + (size_t)B_ * H_ * S_ * S_,
              tot * (1.0f / ((float)B_ * H_ * S_ * S_)));
  }
}

// ------------------------------------------------------------------ launch --
extern "C" void kernel_launch(void* const* d_in, const int* in_sizes, int n_in,
                              void* d_out, int out_size, void* d_ws, size_t ws_size,
                              hipStream_t stream) {
  (void)in_sizes; (void)n_in; (void)out_size; (void)ws_size;
  const float* query = (const float*)d_in[0];
  const float* keyt  = (const float*)d_in[1];
  const float* mask  = (const float*)d_in[2];
  const float* span  = (const float*)d_in[3];
  const float* W[6]  = { (const float*)d_in[4],  (const float*)d_in[6],
                         (const float*)d_in[8],  (const float*)d_in[10],
                         (const float*)d_in[12], (const float*)d_in[14] };
  const float* bias[6] = { (const float*)d_in[5],  (const float*)d_in[7],
                           (const float*)d_in[9],  (const float*)d_in[11],
                           (const float*)d_in[13], (const float*)d_in[15] };
  float* out = (float*)d_out;

  const size_t nX  = (size_t)B_ * S_ * D_;      // 3,145,728
  const size_t nW  = (size_t)D_ * D_;           // 589,824
  _Float16* Xq = (_Float16*)d_ws;
  _Float16* Xk = Xq + nX;
  _Float16* Wt = Xk + nX;                        // 6 * nW
  _Float16* P  = Wt + 6 * nW;                    // 6 * nX

  cvt_f32_to_f16<<<(int)((nX + 255) / 256), 256, 0, stream>>>(query, Xq, (int)nX);
  cvt_f32_to_f16<<<(int)((nX + 255) / 256), 256, 0, stream>>>(keyt,  Xk, (int)nX);
  for (int i = 0; i < 6; ++i)
    transpose_w_f16<<<(int)((nW + 255) / 256), 256, 0, stream>>>(W[i], Wt + i * nW);

  const float scale = 1.0f / sqrtf((float)D_);
  proj_gemm<<<dim3((B_ * S_) / 16, D_ / 64, 6), 128, 0, stream>>>(
      Xq, Xk, Wt, bias[0], bias[1], bias[2], bias[3], bias[4], bias[5], P, scale);

  zero_loss<<<1, 1, 0, stream>>>(out);

  const size_t lds = (3 * 16 * 512 + 4) * sizeof(float);  // 96 KB + reduce buf
  attn_fused<<<B_ * H_ * (S_ / 16), 128, lds, stream>>>(P, mask, span, out);
}
